// DynamicConvolutionAttention_89558658056838
// MI455X (gfx1250) — matrix operand order repaired and checked by reading, not compiled, not run on hardware
//
#include <hip/hip_runtime.h>
#include <hip/hip_bf16.h>

typedef _Float16 f16;
typedef __attribute__((ext_vector_type(16))) _Float16 v16h;
typedef __attribute__((ext_vector_type(8)))  float    v8f;
typedef __attribute__((ext_vector_type(4)))  unsigned int u32x4;
typedef __attribute__((ext_vector_type(8)))  unsigned int u32x8;

#define B_   64
#define T_   2000
#define RNN  1024
#define EMB  512
#define ATT  128
#define SCH  8
#define SK   21
#define DCH  8
#define DK   21
#define PL   11
#define KPAD 34   // padded K stride (halfs); cols 21..33 zero; 68B row -> conflict-free LDS
#define TCH  500  // t-chunk for context partials (4 chunks)

// workspace layout (float-element offsets)
#define WS_G_OFF   0                               // float [B_*DCH*DK]        = 10752
#define WS_E_OFF   (B_*DCH*DK)                     // float [B_*T_]            = 128000
#define WS_K_OFF_F (WS_E_OFF + B_*T_)              // half  [B_*ATT*KPAD]      = 278528 halfs
#define WS_P_OFF   (WS_K_OFF_F + (B_*ATT*KPAD)/2)  // float [B_*4*EMB] partial = 131072

// ---------------- k1: hidden = tanh(q @ W_w^T + W_b); G = hidden @ V_w^T ----------------
__global__ __launch_bounds__(256) void k1_hidden_G(
    const float* __restrict__ q, const float* __restrict__ Ww,
    const float* __restrict__ Wb, const float* __restrict__ Vw,
    float* __restrict__ Gout)
{
  __shared__ float sH[64 * ATT];          // 32 KB
  const int tid  = threadIdx.x;
  const int lane = tid & 31, wv = tid >> 5;
  const int ml = lane & 15, hi = lane >> 4;
  const int m0 = (wv >> 1) * 16;          // 4 M-tiles over 64 rows
  const int nb = (wv & 1) * 4;            // 4 N-tiles per wave over 8 N-tiles

  v8f acc[4];
  #pragma unroll
  for (int j = 0; j < 4; ++j) acc[j] = (v8f){0.f,0.f,0.f,0.f,0.f,0.f,0.f,0.f};

  for (int ks = 0; ks < RNN / 32; ++ks) {
    const int k0 = ks * 32;
    v16h a;                               // A 16x32: lane<16 -> K {0..7,16..23}; hi -> +8
    #pragma unroll
    for (int h = 0; h < 16; ++h) {
      int kk = k0 + h + ((h & 8) ? 8 : 0) + 8 * hi;
      a[h] = (f16)q[(m0 + ml) * RNN + kk];
    }
    #pragma unroll
    for (int j = 0; j < 4; ++j) {
      int n = (nb + j) * 16 + ml;         // B 32x16: lane<16 -> K {0..7,16..23}; hi -> +8
      v16h bv;
      #pragma unroll
      for (int h = 0; h < 16; ++h)
        bv[h] = (f16)Ww[n * RNN + k0 + h + ((h & 8) ? 8 : 0) + 8 * hi];
      acc[j] = __builtin_amdgcn_wmma_f32_16x16x32_f16(false, a, false, bv,
                                                      (short)0, acc[j], false, false);
    }
  }
  #pragma unroll
  for (int j = 0; j < 4; ++j) {           // C: lane n = ml, row m = r + 8*hi
    int n = (nb + j) * 16 + ml;
    float wb = Wb[n];
    #pragma unroll
    for (int r = 0; r < 8; ++r)
      sH[(m0 + r + 8 * hi) * ATT + n] = tanhf(acc[j][r] + wb);
  }
  __syncthreads();

  for (int o = tid; o < B_ * DCH * DK; o += 256) {   // G = hidden @ V_w^T (64x168, K=128)
    int bi = o / (DCH * DK), jj = o % (DCH * DK);
    float s = 0.f;
    #pragma unroll 4
    for (int a2 = 0; a2 < ATT; ++a2)
      s += sH[bi * ATT + a2] * Vw[jj * ATT + a2];
    Gout[o] = s;
  }
}

// ---------------- k2: per-batch fused conv kernel  K_b[a][k] (128 x 21, padded) ----------
__global__ __launch_bounds__(128) void k2_buildK(
    const float* __restrict__ G, const float* __restrict__ Fw,
    const float* __restrict__ Uw, const float* __restrict__ Tw,
    f16* __restrict__ Kp)
{
  const int b = blockIdx.x, a = threadIdx.x;           // a = 0..127
  const float* g = G + b * DCH * DK;
  for (int k = 0; k < KPAD; ++k) {
    float v = 0.f;
    if (k < DK) {
      #pragma unroll
      for (int c = 0; c < SCH; ++c)
        v += Uw[a * SCH + c] * Fw[c * SK + k] + Tw[a * DCH + c] * g[c * DK + k];
    }
    Kp[(b * ATT + a) * KPAD + k] = (f16)v;
  }
}

// ---------------- k3: e[b,t] = v . tanh(A_b K_b^T + T_b) + log(prior)  (WMMA + TDM) ------
__global__ __launch_bounds__(256) void k3_escore(
    const float* __restrict__ ap, const f16* __restrict__ Kp,
    const float* __restrict__ vw, const float* __restrict__ Tb,
    const float* __restrict__ P, float* __restrict__ e)
{
  __shared__ f16   sK[ATT * KPAD];   // 8.5 KB, filled by Tensor Data Mover
  __shared__ float sV[ATT], sT[ATT];
  const int b = blockIdx.x, tid = threadIdx.x;
  for (int i = tid; i < ATT; i += 256) { sV[i] = vw[i]; sT[i] = Tb[i]; }

  if (tid < 32) {
    // ---- TDM: 1-D descriptor copy of K_b (2176 dwords) global -> LDS ----
    unsigned long long ga = (unsigned long long)(const void*)(Kp + (size_t)b * ATT * KPAD);
    unsigned lds = (unsigned)(unsigned long long)(void*)&sK[0];   // LDS byte offset
    const unsigned NW = (ATT * KPAD) / 2;                         // 2176 dwords
    u32x4 g0;
    g0[0] = 1u;                                        // count=1 (valid), user mode
    g0[1] = lds;                                       // lds_addr
    g0[2] = (unsigned)(ga & 0xFFFFFFFFu);              // global_addr[31:0]
    g0[3] = (unsigned)((ga >> 32) & 0x01FFFFFFu) | (2u << 30);  // ga[56:32] | type=2
    u32x8 g1;
    g1[0] = (2u << 16);                                // data_size = 4B; no multicast/pad
    g1[1] = (NW & 0xFFFFu) << 16;                      // tensor_dim0[15:0]
    g1[2] = ((NW >> 16) & 0xFFFFu) | (1u << 16);       // tensor_dim0[31:16], tensor_dim1=1
    g1[3] = (NW & 0xFFFFu) << 16;                      // tile_dim0 = 2176
    g1[4] = 0u;                                        // tile_dim1=0 (1-D), tile_dim2=0
    g1[5] = NW;                                        // tensor_dim0_stride lo32
    g1[6] = 0u;
    g1[7] = 0u;
    asm volatile("tensor_load_to_lds %0, %1" :: "s"(g0), "s"(g1) : "memory");
    __builtin_amdgcn_s_wait_tensorcnt(0);
  }
  __syncthreads();

  const int wv = tid >> 5, lane = tid & 31, ml = lane & 15, hi = lane >> 4;
  const int tile = blockIdx.y * 8 + wv;                 // 125 tiles of 16 timesteps
  if (tile >= T_ / 16) return;                          // whole-wave uniform branch
  const int t0 = tile * 16;
  const float* apb = ap + b * T_;

  const int tm = t0 + ml;                               // A 16x32 sliding-window tile
  v16h a;
  #pragma unroll
  for (int h = 0; h < 16; ++h) {
    int K = h + ((h & 8) ? 8 : 0) + 8 * hi;
    int idx = tm + K - 10;
    float v = (K < DK && idx >= 0 && idx < T_) ? apb[idx] : 0.f;
    a[h] = (f16)v;
  }

  float part[8];
  #pragma unroll
  for (int r = 0; r < 8; ++r) part[r] = 0.f;

  #pragma unroll
  for (int nt = 0; nt < 8; ++nt) {
    const int n = nt * 16 + ml;
    v16h bv;
    #pragma unroll
    for (int h = 0; h < 16; ++h)
      bv[h] = sK[n * KPAD + h + ((h & 8) ? 8 : 0) + 8 * hi];
    v8f c = (v8f){0.f,0.f,0.f,0.f,0.f,0.f,0.f,0.f};
    c = __builtin_amdgcn_wmma_f32_16x16x32_f16(false, a, false, bv,
                                               (short)0, c, false, false);
    const float vb = sV[n], tb = sT[n];
    #pragma unroll
    for (int r = 0; r < 8; ++r)
      part[r] += vb * tanhf(c[r] + tb);      // lane holds (m=r+8*hi, n)
  }
  #pragma unroll
  for (int r = 0; r < 8; ++r)                // reduce across 16-lane N groups
    for (int mk = 8; mk >= 1; mk >>= 1)
      part[r] += __shfl_xor(part[r], mk, 32);

  if (ml == 0) {                             // lanes 0 and 16 write 8 rows each
    #pragma unroll
    for (int r = 0; r < 8; ++r) {
      int t = t0 + 8 * hi + r;
      float pr = 0.f;
      #pragma unroll
      for (int j = 0; j < PL; ++j) {         // causal prior conv, left zero-pad
        int idx = t - 10 + j;
        if (idx >= 0) pr += apb[idx] * P[j];
      }
      e[b * T_ + t] = part[r] + logf(fmaxf(pr, 1e-6f));
    }
  }
}

// ---------------- k4: row softmax over T with mask --------------------------------------
__global__ __launch_bounds__(256) void k4_softmax(
    const float* __restrict__ e, const unsigned char* __restrict__ mask,
    float* __restrict__ wout)
{
  __shared__ float red[256];
  const int b = blockIdx.x, tid = threadIdx.x;
  const float* eb = e + b * T_;
  const unsigned char* mb = mask + b * T_;

  float mx = -__builtin_inff();
  for (int t = tid; t < T_; t += 256)
    mx = fmaxf(mx, mb[t] ? -__builtin_inff() : eb[t]);
  red[tid] = mx; __syncthreads();
  for (int s = 128; s > 0; s >>= 1) {
    if (tid < s) red[tid] = fmaxf(red[tid], red[tid + s]);
    __syncthreads();
  }
  mx = red[0]; __syncthreads();

  float sm = 0.f;
  for (int t = tid; t < T_; t += 256)
    sm += expf((mb[t] ? -__builtin_inff() : eb[t]) - mx);
  red[tid] = sm; __syncthreads();
  for (int s = 128; s > 0; s >>= 1) {
    if (tid < s) red[tid] += red[tid + s];
    __syncthreads();
  }
  const float inv = 1.f / red[0];
  for (int t = tid; t < T_; t += 256)
    wout[b * T_ + t] = expf((mb[t] ? -__builtin_inff() : eb[t]) - mx) * inv;
}

// ---------------- k5: context partials, float4 streams of memory (262 MB) ----------------
__global__ __launch_bounds__(128) void k5_context_part(
    const float* __restrict__ w, const float* __restrict__ mem,
    float* __restrict__ part)
{
  __shared__ float sW[TCH];                 // 2 KB
  const int b = blockIdx.x, chunk = blockIdx.y;
  const int t0 = chunk * TCH;
  for (int t = threadIdx.x; t < TCH; t += 128) sW[t] = w[b * T_ + t0 + t];
  __syncthreads();
  const float4* m4 = (const float4*)(mem + (size_t)b * T_ * EMB) + (size_t)t0 * (EMB / 4)
                     + threadIdx.x;         // lane -> 4 channels, b128 loads
  float4 acc; acc.x = acc.y = acc.z = acc.w = 0.f;
  #pragma unroll 4
  for (int t = 0; t < TCH; ++t) {
    float4 v = m4[(size_t)t * (EMB / 4)];   // 512B per wave per load
    float s = sW[t];
    acc.x += s * v.x; acc.y += s * v.y; acc.z += s * v.z; acc.w += s * v.w;
  }
  ((float4*)part)[(size_t)(b * 4 + chunk) * (EMB / 4) + threadIdx.x] = acc;
}

__global__ __launch_bounds__(256) void k6_context_reduce(
    const float* __restrict__ part, float* __restrict__ ctx)
{
  const int i = blockIdx.x * 256 + threadIdx.x;        // 64*512 outputs
  const int b = i / EMB, ch = i % EMB;
  float s = 0.f;
  #pragma unroll
  for (int c = 0; c < 4; ++c)
    s += part[(size_t)(b * 4 + c) * EMB + ch];
  ctx[i] = s;
}

extern "C" void kernel_launch(void* const* d_in, const int* in_sizes, int n_in,
                              void* d_out, int out_size, void* d_ws, size_t ws_size,
                              hipStream_t stream) {
  const float* q    = (const float*)d_in[0];   // (64,1024)
  const float* mem  = (const float*)d_in[1];   // (64,2000,512)
  const float* ap   = (const float*)d_in[2];   // (64,2000)
  const unsigned char* mask = (const unsigned char*)d_in[3];  // (64,2000) bool
  const float* Ww   = (const float*)d_in[4];   // (128,1024)
  const float* Wb   = (const float*)d_in[5];   // (128)
  const float* Vw   = (const float*)d_in[6];   // (168,128)
  const float* Fw   = (const float*)d_in[7];   // (8,1,21)
  const float* Uw   = (const float*)d_in[8];   // (128,8)
  const float* Tw   = (const float*)d_in[9];   // (128,8)
  const float* Tb   = (const float*)d_in[10];  // (128)
  const float* vw   = (const float*)d_in[11];  // (1,128)
  const float* P    = (const float*)d_in[12];  // (11)

  float* out = (float*)d_out;
  float* ctx = out;                 // (64,512) first
  float* wts = out + B_ * EMB;      // (64,2000) second

  float* ws = (float*)d_ws;
  float* G    = ws + WS_G_OFF;
  float* e    = ws + WS_E_OFF;
  f16*   Kp   = (f16*)(ws + WS_K_OFF_F);
  float* prt  = ws + WS_P_OFF;

  k1_hidden_G<<<1, 256, 0, stream>>>(q, Ww, Wb, Vw, G);
  k2_buildK <<<B_, 128, 0, stream>>>(G, Fw, Uw, Tw, Kp);
  k3_escore <<<dim3(B_, 16), 256, 0, stream>>>(ap, Kp, vw, Tb, P, e);
  k4_softmax<<<B_, 256, 0, stream>>>(e, mask, wts);
  k5_context_part<<<dim3(B_, T_ / TCH), 128, 0, stream>>>(wts, mem, prt);
  k6_context_reduce<<<(B_ * EMB) / 256, 256, 0, stream>>>(prt, ctx);
}
